// LSTM_68358699483702
// MI455X (gfx1250) — compile-verified, matrix operand order checked by reading, not run on hardware
//
#include <hip/hip_runtime.h>
#include <hip/hip_bf16.h>
#include <stdint.h>

// ---------------------------------------------------------------------------
// LSTM encoder/decoder on MI455X (gfx1250), persistent-kernel design.
//   B=128, T=1024, I=64, H=256, C=64, OUT_LEN=128
// Per step: gates(128x1024) = [x_t | h] (128x320) @ [W_ih|W_hh]^T, bf16 WMMA,
// f32 accumulate. 16 blocks (one 16-wide H slice each), 8 wave32 per block.
// Weights live in LDS, h/c live in registers + bf16 ping-pong buffers in L2.
// ---------------------------------------------------------------------------

#define DEV __device__ __forceinline__

constexpr int B_   = 128;
constexpr int T_   = 1024;
constexpr int I_   = 64;
constexpr int H_   = 256;
constexpr int C_   = 64;
constexpr int OUTL = 128;
constexpr int KA   = I_ + H_;   // 320 : fused [x | h] K-dimension
constexpr int NBLK = 16;        // == max workgroup-cluster size on CDNA5
constexpr int NTHR = 256;       // 8 wave32

typedef __attribute__((ext_vector_type(16))) __bf16 v16bf;
typedef __attribute__((ext_vector_type(8)))  float  v8f;

union FragU { v16bf v; uint4 q[2]; };

// A/B fragment loader for V_WMMA_F32_16X16X32_BF16 (ISA 7.12.2, 16-bit 16x32):
// lane<16 : row r=lane, K = [k0..k0+7] then [k0+16..k0+23]
// lane>=16: row r=lane-16, K = [k0+8..k0+15] then [k0+24..k0+31]
// Works for A (row=M of row-major A) and for B (row=N of row-major B^T).
DEV v16bf load_frag(const __bf16* base, int ld, int lane) {
  const int r  = lane & 15;
  const int hf = lane >> 4;
  const __bf16* p = base + r * ld + hf * 8;
  FragU f;
  f.q[0] = *(const uint4*)(p);
  f.q[1] = *(const uint4*)(p + 16);
  return f.v;
}

DEV v8f wmma_bf16(v16bf a, v16bf b, v8f c) {
  // (neg_a, A, neg_b, B, c_mod, C, reuse_a, reuse_b)
  return __builtin_amdgcn_wmma_f32_16x16x32_bf16(false, a, false, b,
                                                 (short)0, c, false, false);
}

DEV float sigm(float x) {
  return __builtin_amdgcn_rcpf(1.0f + __expf(-x));
}
DEV float tanh_(float x) {
  x = fminf(fmaxf(x, -15.0f), 15.0f);
  float e = __expf(2.0f * x);
  return (e - 1.0f) * __builtin_amdgcn_rcpf(e + 1.0f);
}

// Device-wide sense barrier over NBLK resident blocks. bar[0]=count, bar[1]=gen
DEV void grid_barrier(unsigned* bar) {
  __threadfence();          // make this thread's global stores device-visible
  __syncthreads();
  if (threadIdx.x == 0) {
    unsigned g0 = __hip_atomic_load(&bar[1], __ATOMIC_ACQUIRE,
                                    __HIP_MEMORY_SCOPE_AGENT);
    unsigned a  = __hip_atomic_fetch_add(&bar[0], 1u, __ATOMIC_ACQ_REL,
                                         __HIP_MEMORY_SCOPE_AGENT);
    if (a == (unsigned)(NBLK - 1)) {
      __hip_atomic_store(&bar[0], 0u, __ATOMIC_RELAXED,
                         __HIP_MEMORY_SCOPE_AGENT);
      __hip_atomic_fetch_add(&bar[1], 1u, __ATOMIC_RELEASE,
                             __HIP_MEMORY_SCOPE_AGENT);
    } else {
      while (__hip_atomic_load(&bar[1], __ATOMIC_ACQUIRE,
                               __HIP_MEMORY_SCOPE_AGENT) == g0)
        __builtin_amdgcn_s_sleep(1);
    }
  }
  __syncthreads();
}

// One-time x : f32 -> bf16 (keeps recurrent A-fragment loads as pure b128)
__global__ void cvt_bf16_kernel(const float* __restrict__ src,
                                __bf16* __restrict__ dst, int n) {
  for (int i = blockIdx.x * blockDim.x + threadIdx.x; i < n;
       i += gridDim.x * blockDim.x)
    dst[i] = (__bf16)src[i];
}

__global__ void __launch_bounds__(NTHR, 1)
lstm_persistent(const int*   __restrict__ lengths,
                const float* __restrict__ W_ih,  // (4H, I)
                const float* __restrict__ W_hh,  // (4H, H)
                const float* __restrict__ b_ih,
                const float* __restrict__ b_hh,
                const float* __restrict__ W1,    // (H, H)
                const float* __restrict__ b1,
                const float* __restrict__ W2,    // (C, H)
                const float* __restrict__ b2,
                const __bf16* __restrict__ xb,   // (B, T, I) bf16
                __bf16* __restrict__ hb,         // 2 x (B, H) ping-pong
                __bf16* __restrict__ ub,         // 2 x (B, H) ping-pong
                unsigned* __restrict__ bar,
                float* __restrict__ out)         // (B, OUT_LEN, C)
{
  // Per-block weight slices, bf16, resident in LDS for the whole kernel.
  __shared__ __bf16 sW [64 * KA];   // [n=gate-col local 0..63][k 0..319] 40 KB
  __shared__ __bf16 sW1[16 * H_];   // [n=u-col local 0..15 ][k 0..255]   8 KB
  __shared__ __bf16 sW2[C_ * H_];   // [n=class 0..63      ][k 0..255]  32 KB

  const int tid   = threadIdx.x;
  const int blk   = blockIdx.x;
  const int hcol0 = blk * 16;               // this block's H slice
  const int wave  = tid >> 5;
  const int lane  = tid & 31;
  const int cn    = lane & 15;              // tile column (N) this lane owns
  const int hf    = lane >> 4;              // accumulator row-half
  const int mt    = wave;                   // M-tile (batch rows 16*mt..)

  // ---- stage weights into LDS (f32 -> bf16) -------------------------------
  for (int idx = tid; idx < 64 * KA; idx += NTHR) {
    int n = idx / KA, k = idx % KA;
    int g = n >> 4, c = n & 15;
    int row = g * H_ + hcol0 + c;           // gate row in (4H, *)
    float v = (k < I_) ? W_ih[row * I_ + k] : W_hh[row * H_ + (k - I_)];
    sW[n * KA + k] = (__bf16)v;
  }
  for (int idx = tid; idx < 16 * H_; idx += NTHR) {
    int n = idx / H_, k = idx % H_;
    sW1[n * H_ + k] = (__bf16)W1[(hcol0 + n) * H_ + k];
  }
  for (int idx = tid; idx < C_ * H_; idx += NTHR)
    sW2[idx] = (__bf16)W2[idx];

  // ---- zero both h ping-pong buffers (cooperative across grid) ------------
  for (int i = blk * NTHR + tid; i < 2 * B_ * H_; i += NBLK * NTHR)
    hb[i] = (__bf16)0.0f;

  // ---- per-lane constant state --------------------------------------------
  const int b0 = mt * 16 + hf * 8;          // first of this lane's 8 batch rows
  int   lenv[8];
  float biasv[4];
  v8f   c_reg, h_reg;
#pragma unroll
  for (int r = 0; r < 8; ++r) {
    lenv[r]  = lengths[b0 + r];
    c_reg[r] = 0.0f;
    h_reg[r] = 0.0f;
  }
#pragma unroll
  for (int g = 0; g < 4; ++g) {
    int col = g * H_ + hcol0 + cn;
    biasv[g] = b_ih[col] + b_hh[col];
  }
  const float b1v = b1[hcol0 + cn];
  float b2v = 0.0f;
  int   omt = 0, ont = 0;
  if (wave < 2) {                            // out-GEMM tile assignment
    int gt = blk * 2 + wave;                 // 0..31 over 8x4 tiles
    omt = gt >> 2; ont = gt & 3;
    b2v = b2[ont * 16 + cn];
  }

  grid_barrier(bar);

  // =========================================================================
  // 1152 dependent steps: 1024 encoder (masked) + 128 decoder
  // =========================================================================
  for (int s = 0; s < T_ + OUTL; ++s) {
    const bool enc = (s < T_);
    const __bf16* hcur  = hb + (size_t)(s & 1) * B_ * H_;
    __bf16*       hnext = hb + (size_t)((s + 1) & 1) * B_ * H_;

    // ---- gates = [x_t | h] @ [W_ih | W_hh]^T  (+bias later) ---------------
    v8f aI = {0,0,0,0,0,0,0,0}, aF = aI, aG = aI, aO = aI;
#pragma unroll
    for (int kt = 0; kt < KA / 32; ++kt) {   // 10 K-tiles of 32
      if (!enc && kt < I_ / 32) continue;    // decoder: no x contribution
      const int k = kt * 32;
      v16bf afrag;
      if (k < I_)
        afrag = load_frag(xb + ((size_t)b0 - (size_t)(hf * 8)) * 0 /*dummy*/ +
                              (size_t)(mt * 16) * T_ * I_ + (size_t)s * I_ + k,
                          T_ * I_, lane);
      else
        afrag = load_frag(hcur + (size_t)(mt * 16) * H_ + (k - I_), H_, lane);
      aI = wmma_bf16(afrag, load_frag(sW + (0 * 16) * KA + k, KA, lane), aI);
      aF = wmma_bf16(afrag, load_frag(sW + (1 * 16) * KA + k, KA, lane), aF);
      aG = wmma_bf16(afrag, load_frag(sW + (2 * 16) * KA + k, KA, lane), aG);
      aO = wmma_bf16(afrag, load_frag(sW + (3 * 16) * KA + k, KA, lane), aO);
    }
    if (enc && s + 1 < T_)                    // overlap next x tile fetch
      __builtin_prefetch(xb + ((size_t)b0 * T_ + (s + 1)) * I_, 0, 1);

    // ---- LSTM cell (lane-local: i/f/g/o of (b,col) live in this lane) -----
#pragma unroll
    for (int r = 0; r < 8; ++r) {
      float ig = sigm (aI[r] + biasv[0]);
      float fg = sigm (aF[r] + biasv[1]);
      float gg = tanh_(aG[r] + biasv[2]);
      float og = sigm (aO[r] + biasv[3]);
      float c2 = fg * c_reg[r] + ig * gg;
      float h2 = og * tanh_(c2);
      if (enc) {
        float m = (s < lenv[r]) ? 1.0f : 0.0f;
        c2 = m * c2 + (1.0f - m) * c_reg[r];
        h2 = m * h2 + (1.0f - m) * h_reg[r];
      }
      c_reg[r] = c2;
      h_reg[r] = h2;
      hnext[(size_t)(b0 + r) * H_ + hcol0 + cn] = (__bf16)h2;
    }

    grid_barrier(bar);   // h(s+1) complete & visible everywhere

    if (enc) continue;

    // =======================================================================
    // Decoder head for step t = s - T_:
    //   u   = relu(h2 @ W1^T + b1)      (block: 16 u-cols, wave: 1 M-tile)
    //   out = u @ W2^T + b2             (32 tiles, 2 per block on waves 0-1)
    // =======================================================================
    const int t = s - T_;
    __bf16* unext = ub + (size_t)(s & 1) * B_ * H_;

    {
      v8f au = {0,0,0,0,0,0,0,0};
#pragma unroll
      for (int kt = 0; kt < H_ / 32; ++kt) {
        v16bf afrag = load_frag(hnext + (size_t)(mt * 16) * H_ + kt * 32,
                                H_, lane);
        v16bf bfrag = load_frag(sW1 + kt * 32, H_, lane);
        au = wmma_bf16(afrag, bfrag, au);
      }
#pragma unroll
      for (int r = 0; r < 8; ++r) {
        float u = au[r] + b1v;
        u = (u > 0.0f) ? u : 0.0f;
        unext[(size_t)(b0 + r) * H_ + hcol0 + cn] = (__bf16)u;
      }
    }

    grid_barrier(bar);   // u complete & visible everywhere

    if (wave < 2) {
      v8f ao = {0,0,0,0,0,0,0,0};
#pragma unroll
      for (int kt = 0; kt < H_ / 32; ++kt) {
        v16bf afrag = load_frag(unext + (size_t)(omt * 16) * H_ + kt * 32,
                                H_, lane);
        v16bf bfrag = load_frag(sW2 + (size_t)(ont * 16) * H_ + kt * 32,
                                H_, lane);
        ao = wmma_bf16(afrag, bfrag, ao);
      }
      const int ob0 = omt * 16 + hf * 8;
#pragma unroll
      for (int r = 0; r < 8; ++r)
        out[((size_t)(ob0 + r) * OUTL + t) * C_ + ont * 16 + cn] = ao[r] + b2v;
    }
    // out-GEMM reads ub[s&1]; next step's u-phase writes ub[(s+1)&1] and is
    // barrier-ordered after this block finishes here -> no extra barrier.
  }
}

// ---------------------------------------------------------------------------
// Host launcher.  Workspace layout (needs ~17.0 MB):
//   [0,256)                : barrier {count, gen}
//   [256, +2*B*H*2)        : h ping-pong (bf16)
//   [.., +2*B*H*2)         : u ping-pong (bf16)
//   [.., +B*T*I*2)         : x in bf16
// ---------------------------------------------------------------------------
extern "C" void kernel_launch(void* const* d_in, const int* in_sizes, int n_in,
                              void* d_out, int out_size, void* d_ws,
                              size_t ws_size, hipStream_t stream) {
  const float* x       = (const float*)d_in[0];
  const int*   lengths = (const int*)  d_in[1];
  // d_in[2] = out_lengths scalar (==OUTL), compile-time constant here
  const float* W_ih = (const float*)d_in[3];
  const float* W_hh = (const float*)d_in[4];
  const float* b_ih = (const float*)d_in[5];
  const float* b_hh = (const float*)d_in[6];
  const float* W1   = (const float*)d_in[7];
  const float* b1   = (const float*)d_in[8];
  const float* W2   = (const float*)d_in[9];
  const float* b2   = (const float*)d_in[10];
  float* out = (float*)d_out;

  char* ws = (char*)d_ws;
  unsigned* bar = (unsigned*)ws;
  size_t off = 256;
  __bf16* hb = (__bf16*)(ws + off); off += (size_t)2 * B_ * H_ * 2;
  __bf16* ub = (__bf16*)(ws + off); off += (size_t)2 * B_ * H_ * 2;
  __bf16* xb = (__bf16*)(ws + off);

  hipMemsetAsync(bar, 0, 256, stream);

  const int nx = B_ * T_ * I_;
  cvt_bf16_kernel<<<2048, 256, 0, stream>>>(x, xb, nx);

  lstm_persistent<<<NBLK, NTHR, 0, stream>>>(lengths, W_ih, W_hh, b_ih, b_hh,
                                             W1, b1, W2, b2, xb, hb, ub, bar,
                                             out);
}
